// NeuralTPP_52355651338552
// MI455X (gfx1250) — matrix operand order repaired and checked by reading, not compiled, not run on hardware
//
#include <hip/hip_runtime.h>
#include <hip/hip_bf16.h>

#define NB    4096
#define LL    512
#define HH_   32
#define EPSF  1e-8f
#define NBLK  (NB / 16)

typedef __attribute__((ext_vector_type(16))) __bf16 v16bf;
typedef __attribute__((ext_vector_type(8)))  float  v8f;

__device__ __forceinline__ float fast_rcp(float x) {
    return __builtin_amdgcn_rcpf(x);
}
// CDNA5 has a hardware V_TANH_F32 (ISA §8: TRANS ops = SIN/COS/SQRT/RCP/RSQ/TANH/EXP/LOG)
__device__ __forceinline__ float hw_tanh(float x) {
#if __has_builtin(__builtin_amdgcn_tanhf)
    return __builtin_amdgcn_tanhf(x);
#elif __has_builtin(__builtin_amdgcn_tanh_f32)
    return __builtin_amdgcn_tanh_f32(x);
#else
    return 2.0f * fast_rcp(1.0f + __expf(-2.0f * x)) - 1.0f;
#endif
}
// sigmoid(x) = 0.5*tanh(x/2) + 0.5  (single TRANS op)
__device__ __forceinline__ float fast_sigmoid(float x) {
    return fmaf(hw_tanh(0.5f * x), 0.5f, 0.5f);
}
__device__ __forceinline__ float fast_softplus(float x) {
    return fmaxf(x, 0.0f) + __logf(1.0f + __expf(-fabsf(x)));
}

// XOR-swap within a 16-lane row via DPP16 ROW_XMASK (pure VALU, no LDS)
template <int M>
__device__ __forceinline__ float row_xor(float x) {
    return __int_as_float(__builtin_amdgcn_update_dpp(
        0, __float_as_int(x), 0x160 | M, 0xf, 0xf, true));
}
__device__ __forceinline__ float row_reduce16(float x) {
    x += row_xor<1>(x);
    x += row_xor<2>(x);
    x += row_xor<4>(x);
    x += row_xor<8>(x);
    return x;   // all 16 lanes of the row hold the sum
}

__device__ __forceinline__ v8f wmma_bf16(v16bf a, v16bf b, v8f c) {
    return __builtin_amdgcn_wmma_f32_16x16x32_bf16(
        false, a, false, b, (short)0, c, false, false);
}

// One wave (32 lanes) per block; each block owns a 16-row batch tile and runs
// the full 512-step scan. h lives in WMMA C/D-layout f32 accumulators; a 2KB
// LDS tile converts D-layout -> A-layout between steps. Weights stay in VGPRs.
__global__ __launch_bounds__(32)
void tpp_scan_kernel(const float* __restrict__ deltas,
                     const float* __restrict__ mask,
                     const float* __restrict__ w_ih,
                     const float* __restrict__ w_hh,
                     const float* __restrict__ b_ih,
                     const float* __restrict__ b_hh,
                     const float* __restrict__ w1,
                     const float* __restrict__ b1,
                     const float* __restrict__ w2,
                     const float* __restrict__ b2,
                     float* __restrict__ part_tot,
                     float* __restrict__ part_msk)
{
    __shared__ float  hs[16 * HH_];   // h tile, row-major 16 x 32, f32
    __shared__ float2 ts[16];         // per-row (tau, mask) for current step

    const int lane = threadIdx.x;     // 0..31
    const int n    = lane & 15;
    const int hi   = lane >> 4;
    const int rowbase = blockIdx.x * 16;

    // ---------------- B operands (weights), bf16, contiguous-k B layout (same k order as A) -------------
    // B (K x N = 32 x 16): lane holds column n = lane&15;
    // lanes 0-15 hold K=0..15 (element e -> K=e), lanes 16-31 hold K=16..31.
    const int koff = hi * 16;
    v16bf Bw1[2];     // for h @ w1_h.T  (output cols 0..31)
    v16bf Bhh[6];     // for h @ w_hh.T  (output cols 0..95)
#pragma unroll
    for (int t = 0; t < 2; ++t) {
#pragma unroll
        for (int e = 0; e < 16; ++e)
            Bw1[t][e] = (__bf16)w1[(t * 16 + n) * (1 + HH_) + 1 + (koff + e)];
    }
#pragma unroll
    for (int t = 0; t < 6; ++t) {
#pragma unroll
        for (int e = 0; e < 16; ++e)
            Bhh[t][e] = (__bf16)w_hh[(t * 16 + n) * HH_ + (koff + e)];
    }

    // ---------------- per-lane column constants -----------------------------
    const float w1tau0 = w1[n * (1 + HH_)];
    const float w1tau1 = w1[(16 + n) * (1 + HH_)];
    const float b1_0 = b1[n],      b1_1 = b1[16 + n];
    const float w2_0 = w2[n],      w2_1 = w2[16 + n];
    const float ww0  = w1tau0 * w2_0, ww1 = w1tau1 * w2_1;
    const float b2v  = b2[0];
    const float sel0 = (n == 0) ? 1.0f : 0.0f;   // branchless accumulation mask
    float wihc[6], bihc[6], bhhc[6];
#pragma unroll
    for (int t = 0; t < 6; ++t) {
        int c3 = t * 16 + n;
        wihc[t] = w_ih[c3];
        bihc[t] = b_ih[c3];
        bhhc[t] = b_hh[c3];
    }
    // r/z gates allow precombined bias (bias sits outside any product)
    const float brz0 = bihc[0] + bhhc[0];
    const float brz1 = bihc[1] + bhhc[1];
    const float bzz0 = bihc[2] + bhhc[2];
    const float bzz1 = bihc[3] + bhhc[3];

    // ---------------- state init --------------------------------------------
    v8f c0 = {}; v8f c1 = {};        // h cols [0..15], [16..31] in D layout
    for (int i = lane; i < 16 * HH_; i += 32) hs[i] = 0.0f;
    __syncthreads();

    float acc  = 0.0f;   // sum of (log lam - phi) * mask over this tile
    float msum = 0.0f;   // sum of mask over this tile

    // lanes 0-15 stream tau for row n; lanes 16-31 stream mask for row n
    const float* srcrow = (hi ? mask : deltas) + (size_t)(rowbase + n) * LL;
    float* tsf = (float*)ts;          // ts[m] = (tau_m, mask_m)

    for (int t = 0; t < LL; ++t) {
        tsf[(n << 1) | hi] = srcrow[t];
        __builtin_prefetch(srcrow + t + 64, 0, 1);
        __syncthreads();

        // ---- A operand: hs (row-major f32) -> WMMA A layout (16x32 bf16) ----
        const int kbase = hi * 16;
        const float* hp = hs + n * HH_;
        float4 q0 = *(const float4*)(hp + kbase);
        float4 q1 = *(const float4*)(hp + kbase + 4);
        float4 q2 = *(const float4*)(hp + kbase + 8);
        float4 q3 = *(const float4*)(hp + kbase + 12);
        v16bf A;
        A[0]  = (__bf16)q0.x; A[1]  = (__bf16)q0.y; A[2]  = (__bf16)q0.z; A[3]  = (__bf16)q0.w;
        A[4]  = (__bf16)q1.x; A[5]  = (__bf16)q1.y; A[6]  = (__bf16)q1.z; A[7]  = (__bf16)q1.w;
        A[8]  = (__bf16)q2.x; A[9]  = (__bf16)q2.y; A[10] = (__bf16)q2.z; A[11] = (__bf16)q2.w;
        A[12] = (__bf16)q3.x; A[13] = (__bf16)q3.y; A[14] = (__bf16)q3.z; A[15] = (__bf16)q3.w;

        // ---- 8 WMMAs: 16x32 @ 32x32 (w1_h) and 16x32 @ 32x96 (w_hh) --------
        v8f z = {};
        v8f p0 = wmma_bf16(A, Bw1[0], z);
        v8f p1 = wmma_bf16(A, Bw1[1], z);
        v8f g0 = wmma_bf16(A, Bhh[0], z);
        v8f g1 = wmma_bf16(A, Bhh[1], z);
        v8f g2 = wmma_bf16(A, Bhh[2], z);
        v8f g3 = wmma_bf16(A, Bhh[3], z);
        v8f g4 = wmma_bf16(A, Bhh[4], z);
        v8f g5 = wmma_bf16(A, Bhh[5], z);

        // ---- elementwise phase in C/D layout: VGPR r, lane -> (r+8*hi, n) --
#pragma unroll
        for (int r = 0; r < 8; ++r) {
            const int m = r + 8 * hi;
            const float2 tm = ts[m];
            const float tau = tm.x;
            const float mk  = tm.y;

            // intensity MLP
            float pre0 = fmaf(tau, w1tau0, p0[r] + b1_0);
            float pre1 = fmaf(tau, w1tau1, p1[r] + b1_1);
            float a0 = hw_tanh(pre0);
            float a1 = hw_tanh(pre1);
            float rawp = a0 * w2_0 + a1 * w2_1;
            float dpp  = (1.0f - a0 * a0) * ww0 + (1.0f - a1 * a1) * ww1;
            rawp = row_reduce16(rawp);            // DPP row_xmask reduction
            dpp  = row_reduce16(dpp);
            float raw = rawp + b2v;
            float phi = fast_softplus(raw);
            float sg  = fast_sigmoid(raw);
            float dphi = sg * dpp;
            float lam  = fast_softplus(dphi) + EPSF;
            float contrib = (__logf(lam) - phi) * mk;
            acc  = fmaf(sel0, contrib, acc);
            msum = fmaf(sel0, mk, msum);

            // GRU cell (gates r, z, n over col pairs (n, 16+n))
            float rg0 = fast_sigmoid(fmaf(tau, wihc[0], g0[r] + brz0));
            float rg1 = fast_sigmoid(fmaf(tau, wihc[1], g1[r] + brz1));
            float zg0 = fast_sigmoid(fmaf(tau, wihc[2], g2[r] + bzz0));
            float zg1 = fast_sigmoid(fmaf(tau, wihc[3], g3[r] + bzz1));
            // n-gate: tanh(i_n + r * (gh_n + b_hh_n)), i_n = tau*w_ih_n + b_ih_n
            float ng0 = hw_tanh(fmaf(tau, wihc[4], bihc[4]) + rg0 * (g4[r] + bhhc[4]));
            float ng1 = hw_tanh(fmaf(tau, wihc[5], bihc[5]) + rg1 * (g5[r] + bhhc[5]));

            float h0 = fmaf(zg0, c0[r] - ng0, ng0); // (1-z)*n + z*h
            float h1 = fmaf(zg1, c1[r] - ng1, ng1);
            h0 = fmaf(mk, h0 - c0[r], c0[r]);       // masked update
            h1 = fmaf(mk, h1 - c1[r], c1[r]);
            c0[r] = h0;
            c1[r] = h1;
            hs[m * HH_ + n]      = h0;
            hs[m * HH_ + 16 + n] = h1;
        }
        __syncthreads();
    }

    // deterministic wave reduction, lane 0 writes block partials
#pragma unroll
    for (int s = 16; s >= 1; s >>= 1) {
        acc  += __shfl_xor(acc,  s, 32);
        msum += __shfl_xor(msum, s, 32);
    }
    if (lane == 0) {
        part_tot[blockIdx.x] = acc;
        part_msk[blockIdx.x] = msum;
    }
}

// Deterministic final reduction over the 256 block partials.
__global__ __launch_bounds__(32)
void tpp_reduce_kernel(const float* __restrict__ part_tot,
                       const float* __restrict__ part_msk,
                       float* __restrict__ out)
{
    float t = 0.0f, m = 0.0f;
    for (int i = threadIdx.x; i < NBLK; i += 32) {
        t += part_tot[i];
        m += part_msk[i];
    }
#pragma unroll
    for (int s = 16; s >= 1; s >>= 1) {
        t += __shfl_xor(t, s, 32);
        m += __shfl_xor(m, s, 32);
    }
    if (threadIdx.x == 0) out[0] = t / (m + EPSF);
}

extern "C" void kernel_launch(void* const* d_in, const int* in_sizes, int n_in,
                              void* d_out, int out_size, void* d_ws, size_t ws_size,
                              hipStream_t stream) {
    (void)in_sizes; (void)n_in; (void)out_size; (void)ws_size;
    const float* deltas = (const float*)d_in[0];
    const float* mask   = (const float*)d_in[1];
    const float* w_ih   = (const float*)d_in[2];
    const float* w_hh   = (const float*)d_in[3];
    const float* b_ih   = (const float*)d_in[4];
    const float* b_hh   = (const float*)d_in[5];
    const float* w1     = (const float*)d_in[6];
    const float* b1     = (const float*)d_in[7];
    const float* w2     = (const float*)d_in[8];
    const float* b2     = (const float*)d_in[9];

    float* part_tot = (float*)d_ws;
    float* part_msk = part_tot + NBLK;

    tpp_scan_kernel<<<NBLK, 32, 0, stream>>>(deltas, mask, w_ih, w_hh, b_ih, b_hh,
                                             w1, b1, w2, b2, part_tot, part_msk);
    tpp_reduce_kernel<<<1, 32, 0, stream>>>(part_tot, part_msk, (float*)d_out);
}
